// ANETransformerLayer_13589276525254
// MI455X (gfx1250) — compile-verified, matrix operand order checked by reading, not run on hardware
//
#include <hip/hip_runtime.h>
#include <hip/hip_bf16.h>
#include <stdint.h>

#define HIDDEN 2048
#define NUM_K_HEADS 16
#define NUM_V_HEADS 32
#define DK 128
#define DV 128
#define KCONV 4
#define QK_DIM (NUM_K_HEADS*DK)        /* 2048 */
#define V_DIM  (NUM_V_HEADS*DV)        /* 4096 */
#define CONV_INNER (2*QK_DIM + V_DIM)  /* 8192 */
#define BATCH 2
#define SEQ 2048
#define NTOK (BATCH*SEQ)               /* 4096 */

typedef __attribute__((ext_vector_type(16))) __bf16 v16bf;
typedef __attribute__((ext_vector_type(8)))  float  v8f;
typedef __attribute__((ext_vector_type(4)))  unsigned int v4u;
typedef __attribute__((ext_vector_type(8)))  int v8i;
typedef __attribute__((ext_vector_type(4)))  int v4i;

struct __align__(16) U4 { unsigned int x, y, z, w; };
union Frag { v16bf v; U4 q[2]; };

__device__ __forceinline__ unsigned short f2bf(float f) {
  unsigned int u = __float_as_uint(f);
  u += 0x7FFFu + ((u >> 16) & 1u);           // round to nearest even
  return (unsigned short)(u >> 16);
}

// ---------------------------------------------------------------- TDM 2D tile load
// One TENSOR_LOAD_TO_LDS moving a (tile_rows x tile_k) bf16 tile from a row-major
// tensor (stride_elems elems/row) into densely packed LDS. D# per ISA 08 §8.3/8.4.
__device__ __forceinline__ void tdm_load_2d(const unsigned short* gaddr,
                                            unsigned int lds_addr,
                                            unsigned int k_rem,      // tensor_dim0 (elems)
                                            unsigned int rows_rem,   // tensor_dim1 (rows)
                                            unsigned int stride_elems,
                                            unsigned int tile_k,
                                            unsigned int tile_rows) {
  const unsigned long long ga = (unsigned long long)gaddr;
  v4u g0;
  g0[0] = 1u;                                            // count=1, user descriptor
  g0[1] = lds_addr;                                      // lds_addr[31:0]
  g0[2] = (unsigned int)(ga & 0xFFFFFFFFu);              // global_addr[31:0]
  g0[3] = (unsigned int)((ga >> 32) & 0x01FFFFFFu)       // global_addr[56:32]
          | (2u << 30);                                  // type=2 ("image")
  v8i g1;
  g1[0] = (int)(1u << 16);                               // wg_mask=0, data_size=1 (2B)
  g1[1] = (int)((k_rem & 0xFFFFu) << 16);                // tensor_dim0[15:0] @ bits63:48
  g1[2] = (int)(((k_rem >> 16) & 0xFFFFu)                // tensor_dim0[31:16]
          | ((rows_rem & 0xFFFFu) << 16));               // tensor_dim1[15:0]
  g1[3] = (int)(((rows_rem >> 16) & 0xFFFFu)             // tensor_dim1[31:16]
          | ((tile_k & 0xFFFFu) << 16));                 // tile_dim0
  g1[4] = (int)(tile_rows & 0xFFFFu);                    // tile_dim1 (tile_dim2=0)
  g1[5] = (int)stride_elems;                             // tensor_dim0_stride[31:0]
  g1[6] = 0;                                             // stride0[47:32], stride1 lo
  g1[7] = 0;
  const v4i gz4 = {0, 0, 0, 0};
#if __clang_major__ >= 23
  const v8i gz8 = {0, 0, 0, 0, 0, 0, 0, 0};
  __builtin_amdgcn_tensor_load_to_lds(g0, g1, gz4, gz4, gz8, 0);
#else
  __builtin_amdgcn_tensor_load_to_lds(g0, g1, gz4, gz4, 0);
#endif
}

// ---------------------------------------------------------------- cast f32->bf16
__global__ void cast_bf16_kernel(const float* __restrict__ in,
                                 unsigned short* __restrict__ out, int n) {
  int i = blockIdx.x * 256 + threadIdx.x;
  if (i < n) out[i] = f2bf(in[i]);
}

// ---------------------------------------------------------------- bf16 WMMA GEMM
// C[M,N](f32) = A[M,K](bf16 rm) * B[N,K]^T (bf16 rm). 128x128 tile, BK=64,
// 256 threads = 8 waves. Tiles are DMA'd by the Tensor Data Mover into
// double-buffered LDS: wave0 issues the A-tile D#, wave1 the B-tile D#.
#define BM 128
#define BN 128
#define BKT 64

__global__ __launch_bounds__(256)
void gemm_bf16_kernel(const unsigned short* __restrict__ A,
                      const unsigned short* __restrict__ B,
                      float* __restrict__ C, int N, int K) {
  __shared__ __align__(16) unsigned short As[2][BM][BKT];
  __shared__ __align__(16) unsigned short Bs[2][BN][BKT];

  const int tid  = threadIdx.x;
  const int bm   = blockIdx.y * BM;
  const int bn   = blockIdx.x * BN;
  const int wave = tid >> 5;
  const int lane = tid & 31;
  const int wm   = (wave & 3) * 32;   // 4 waves along M, 32 rows each
  const int wn   = (wave >> 2) * 64;  // 2 waves along N, 64 cols each
  const int l16  = lane & 15;
  const int half = lane >> 4;
  const int kbase = half * 8;         // ISA 16-bit A layout: lanes 16-31 start at K=8

  v8f acc[2][4];
#pragma unroll
  for (int i = 0; i < 2; ++i)
#pragma unroll
    for (int j = 0; j < 4; ++j)
#pragma unroll
      for (int r = 0; r < 8; ++r) acc[i][j][r] = 0.0f;

  // prologue: TDM fill of buffer 0
  if (wave == 0)
    tdm_load_2d(A + (size_t)bm * K, (unsigned int)(uintptr_t)&As[0][0][0],
                (unsigned int)K, 0x7FFFFFFFu, (unsigned int)K, BKT, BM);
  else if (wave == 1)
    tdm_load_2d(B + (size_t)bn * K, (unsigned int)(uintptr_t)&Bs[0][0][0],
                (unsigned int)K, (unsigned int)(N - bn), (unsigned int)K, BKT, BN);

  int p = 0;
  for (int k0 = 0; k0 < K; k0 += BKT) {
    __builtin_amdgcn_s_wait_tensorcnt(0);   // no-op for non-issuing waves
    __syncthreads();                         // buffer p complete for all waves
    if (k0 + BKT < K) {                      // prefetch into the other buffer
      if (wave == 0)
        tdm_load_2d(A + (size_t)bm * K + k0 + BKT,
                    (unsigned int)(uintptr_t)&As[p ^ 1][0][0],
                    (unsigned int)(K - k0 - BKT), 0x7FFFFFFFu,
                    (unsigned int)K, BKT, BM);
      else if (wave == 1)
        tdm_load_2d(B + (size_t)bn * K + k0 + BKT,
                    (unsigned int)(uintptr_t)&Bs[p ^ 1][0][0],
                    (unsigned int)(K - k0 - BKT), (unsigned int)(N - bn),
                    (unsigned int)K, BKT, BN);
    }
    // compute: two WMMA K-steps on buffer p
#pragma unroll
    for (int kk = 0; kk < BKT; kk += 32) {
      Frag af[2], bfm[4];
#pragma unroll
      for (int i = 0; i < 2; ++i) {
        const int m = wm + i * 16 + l16;
        af[i].q[0] = *(const U4*)&As[p][m][kk + kbase];
        af[i].q[1] = *(const U4*)&As[p][m][kk + kbase + 16];
      }
#pragma unroll
      for (int j = 0; j < 4; ++j) {
        const int n = wn + j * 16 + l16;
        bfm[j].q[0] = *(const U4*)&Bs[p][n][kk + kbase];
        bfm[j].q[1] = *(const U4*)&Bs[p][n][kk + kbase + 16];
      }
#pragma unroll
      for (int i = 0; i < 2; ++i)
#pragma unroll
        for (int j = 0; j < 4; ++j)
          acc[i][j] = __builtin_amdgcn_wmma_f32_16x16x32_bf16(
              false, af[i].v, false, bfm[j].v, (short)0, acc[i][j], false, false);
    }
    p ^= 1;
  }

  // store C: VGPR r -> row = base + r + 8*(lane>>4), col = base + (lane&15)
#pragma unroll
  for (int i = 0; i < 2; ++i) {
#pragma unroll
    for (int j = 0; j < 4; ++j) {
      const int mrow = bm + wm + i * 16 + half * 8;
      const int ncol = bn + wn + j * 16 + l16;
      if (ncol < N) {
#pragma unroll
        for (int r = 0; r < 8; ++r)
          C[(size_t)(mrow + r) * N + ncol] = acc[i][j][r];
      }
    }
  }
}

// ---------------------------------------------------------------- depthwise conv + SiLU
__global__ void conv_silu_kernel(const float* __restrict__ qkv_raw,
                                 const float* __restrict__ conv_w,
                                 float* __restrict__ qkv_conv,
                                 float* __restrict__ conv_state) {
  const int c = blockIdx.x * 256 + threadIdx.x;   // 0..8191
  const int t = blockIdx.y;                        // 0..4095
  const int b = t >> 11;                           // S = 2048
  const int s = t & (SEQ - 1);
  float acc = 0.0f;
#pragma unroll
  for (int j = 0; j < KCONV; ++j) {
    const int ss = s - (KCONV - 1) + j;
    if (ss >= 0)
      acc += conv_w[c * KCONV + j] *
             qkv_raw[(size_t)(b * SEQ + ss) * CONV_INNER + c];
  }
  const float sil = acc / (1.0f + __expf(-acc));
  qkv_conv[(size_t)t * CONV_INNER + c] = sil;
  if (s >= SEQ - KCONV) {
    const int j = s - (SEQ - KCONV);
    conv_state[((size_t)b * CONV_INNER + c) * KCONV + j] =
        qkv_raw[(size_t)t * CONV_INNER + c];
  }
}

// ---------------------------------------------------------------- l2norm of q,k per head
__global__ __launch_bounds__(128)
void l2norm_kernel(const float* __restrict__ conv,
                   float* __restrict__ qn, float* __restrict__ kn) {
  const int t  = blockIdx.x >> 4;
  const int kh = blockIdx.x & 15;
  const int d  = threadIdx.x;
  const float* base = conv + (size_t)t * CONV_INNER;
  const float qv = base[kh * DK + d];
  const float kv = base[QK_DIM + kh * DK + d];
  __shared__ float rq[128], rk[128];
  rq[d] = qv * qv;
  rk[d] = kv * kv;
  __syncthreads();
  for (int off = 64; off > 0; off >>= 1) {
    if (d < off) { rq[d] += rq[d + off]; rk[d] += rk[d + off]; }
    __syncthreads();
  }
  const float inq = rsqrtf(rq[0] + 1e-6f) * 0.08838834764831845f; // * 1/sqrt(128)
  const float ink = rsqrtf(rk[0] + 1e-6f);
  qn[((size_t)t * NUM_K_HEADS + kh) * DK + d] = qv * inq;
  kn[((size_t)t * NUM_K_HEADS + kh) * DK + d] = kv * ink;
}

// ---------------------------------------------------------------- g / beta
__global__ void gbeta_kernel(const float* __restrict__ ab,
                             const float* __restrict__ A_log,
                             const float* __restrict__ dt_bias,
                             float* __restrict__ g, float* __restrict__ beta) {
  const int idx = blockIdx.x * 256 + threadIdx.x;  // t*32 + h
  const int h = idx & 31;
  const int t = idx >> 5;
  const float a  = ab[(size_t)t * 64 + h];
  const float br = ab[(size_t)t * 64 + 32 + h];
  const float x = a + dt_bias[h];
  const float sp = (x > 20.0f) ? x : log1pf(__expf(x));
  g[idx]    = -__expf(A_log[h]) * sp;
  beta[idx] = 1.0f / (1.0f + __expf(-br));
}

// ---------------------------------------------------------------- sequential gated delta scan
// One block per (b, head); 256 threads as 16x16, state register-resident.
// Waves 0..2 stage step s+1's q/k/v (512B each) into double-buffered LDS with
// GLOBAL_LOAD_ASYNC_TO_LDS_B128 while step s computes.
__global__ __launch_bounds__(256)
void scan_kernel(const float* __restrict__ qn, const float* __restrict__ kn,
                 const float* __restrict__ conv, const float* __restrict__ g,
                 const float* __restrict__ beta, float* __restrict__ y,
                 float* __restrict__ state_out) {
  const int bh = blockIdx.x;
  const int b  = bh >> 5;
  const int h  = bh & 31;
  const int kh = h >> 1;                 // HPG = 2 repeat
  const int tid = threadIdx.x;
  const int tx = tid & 15;               // k block
  const int ty = tid >> 4;               // v block
  const int wave = tid >> 5;
  const int lane = tid & 31;

  float st[8][8];
#pragma unroll
  for (int i = 0; i < 8; ++i)
#pragma unroll
    for (int j = 0; j < 8; ++j) st[i][j] = 0.0f;

  __shared__ float red[128][16];
  __shared__ float delta[128];
  __shared__ __align__(16) float qb[2][128], kb[2][128], vb[2][128];

  // async stage of step s (wave0: q, wave1: k, wave2: v; one b128 per lane)
  auto stage = [&](int buf, int s) {
    const size_t t = (size_t)b * SEQ + s;
    const float* src;
    float* dst;
    if (wave == 0)      { src = qn + (t * NUM_K_HEADS + kh) * DK + lane * 4; dst = &qb[buf][lane * 4]; }
    else if (wave == 1) { src = kn + (t * NUM_K_HEADS + kh) * DK + lane * 4; dst = &kb[buf][lane * 4]; }
    else                { src = conv + t * CONV_INNER + 2 * QK_DIM + h * DV + lane * 4; dst = &vb[buf][lane * 4]; }
    const unsigned int laddr = (unsigned int)(uintptr_t)dst;
    asm volatile("global_load_async_to_lds_b128 %0, %1, off"
                 :: "v"(laddr), "v"(src) : "memory");
  };

  // prologue: stage step 0, plus scalar g/beta prefetch
  if (wave < 3) stage(0, 0);
  float gcur = g[(size_t)b * SEQ * NUM_V_HEADS + h];
  float bcur = beta[(size_t)b * SEQ * NUM_V_HEADS + h];
  asm volatile("s_wait_asynccnt 0x0" ::: "memory");
  __syncthreads();

  int p = 0;
  for (int s = 0; s < SEQ; ++s) {
    const size_t t = (size_t)b * SEQ + s;
    float gnxt = 0.0f, bnxt = 0.0f;
    if (s + 1 < SEQ) {
      gnxt = g[(t + 1) * NUM_V_HEADS + h];
      bnxt = beta[(t + 1) * NUM_V_HEADS + h];
    }
    // step operands from LDS buffer p
    float q8[8], k8[8];
#pragma unroll
    for (int i = 0; i < 8; ++i) { q8[i] = qb[p][tx * 8 + i]; k8[i] = kb[p][tx * 8 + i]; }
    const float decay = __expf(gcur);

    // state *= exp(g); partial kv_mem
#pragma unroll
    for (int vi = 0; vi < 8; ++vi) {
      float acc = 0.0f;
#pragma unroll
      for (int ki = 0; ki < 8; ++ki) {
        st[vi][ki] *= decay;
        acc += st[vi][ki] * k8[ki];
      }
      red[ty * 8 + vi][tx] = acc;
    }
    __syncthreads();
    // stage step s+1 into the other buffer (last read two barriers ago)
    if (wave < 3 && s + 1 < SEQ) stage(p ^ 1, s + 1);
    if (tid < 128) {
      float sum = 0.0f;
#pragma unroll
      for (int j = 0; j < 16; ++j) sum += red[tid][j];
      delta[tid] = (vb[p][tid] - sum) * bcur;
    }
    __syncthreads();

    // state += k (x) delta ; partial o = state . q
#pragma unroll
    for (int vi = 0; vi < 8; ++vi) {
      const float dl = delta[ty * 8 + vi];
      float acc = 0.0f;
#pragma unroll
      for (int ki = 0; ki < 8; ++ki) {
        st[vi][ki] += k8[ki] * dl;
        acc += st[vi][ki] * q8[ki];
      }
      red[ty * 8 + vi][tx] = acc;
    }
    asm volatile("s_wait_asynccnt 0x0" ::: "memory"); // staged buffer landed
    __syncthreads();
    if (tid < 128) {
      float o = 0.0f;
#pragma unroll
      for (int j = 0; j < 16; ++j) o += red[tid][j];
      y[(t * NUM_V_HEADS + h) * DV + tid] = o;
    }
    __syncthreads();   // next step may now read buffer p^1 and rewrite red

    gcur = gnxt; bcur = bnxt;
    p ^= 1;
  }

  float* sp = state_out + (size_t)bh * DV * DK;
#pragma unroll
  for (int vi = 0; vi < 8; ++vi)
#pragma unroll
    for (int ki = 0; ki < 8; ++ki)
      sp[(ty * 8 + vi) * DK + tx * 8 + ki] = st[vi][ki];
}

// ---------------------------------------------------------------- RMS norm * norm_w * silu(z) -> bf16
__global__ __launch_bounds__(128)
void gate_kernel(const float* __restrict__ y, const float* __restrict__ z,
                 const float* __restrict__ norm_w,
                 unsigned short* __restrict__ yout) {
  const int th = blockIdx.x;        // t*32 + h
  const int t = th >> 5;
  const int h = th & 31;
  const int d = threadIdx.x;
  const float yv = y[(size_t)th * DV + d];
  __shared__ float r[128];
  r[d] = yv * yv;
  __syncthreads();
  for (int off = 64; off > 0; off >>= 1) {
    if (d < off) r[d] += r[d + off];
    __syncthreads();
  }
  const float rn = rsqrtf(r[0] * (1.0f / 128.0f) + 1e-6f);
  const float zv = z[(size_t)t * V_DIM + h * DV + d];
  const float sz = zv / (1.0f + __expf(-zv));
  yout[(size_t)t * V_DIM + h * DV + d] = f2bf(norm_w[d] * yv * rn * sz);
}

// ---------------------------------------------------------------- launch
extern "C" void kernel_launch(void* const* d_in, const int* in_sizes, int n_in,
                              void* d_out, int out_size, void* d_ws, size_t ws_size,
                              hipStream_t stream) {
  const float* hidden  = (const float*)d_in[0];
  const float* W_qkv   = (const float*)d_in[1];
  const float* W_z     = (const float*)d_in[2];
  const float* W_a     = (const float*)d_in[3];
  const float* W_b     = (const float*)d_in[4];
  const float* conv_w  = (const float*)d_in[5];
  const float* A_log   = (const float*)d_in[6];
  const float* dt_bias = (const float*)d_in[7];
  const float* norm_w  = (const float*)d_in[8];
  const float* W_out   = (const float*)d_in[9];

  float* out        = (float*)d_out;                                   // (B,S,H)
  float* conv_state = out + (size_t)NTOK * HIDDEN;                     // (B,8192,4)
  float* state_out  = conv_state + (size_t)BATCH * CONV_INNER * KCONV; // (B,32,128,128)

  char* ws = (char*)d_ws;
  size_t cur = 0;
  auto alloc = [&](size_t bytes) {
    void* p = ws + cur;
    cur += (bytes + 255) & ~(size_t)255;
    return p;
  };
  unsigned short* h_bf    = (unsigned short*)alloc((size_t)NTOK * HIDDEN * 2);
  unsigned short* wqkv_bf = (unsigned short*)alloc((size_t)CONV_INNER * HIDDEN * 2);
  unsigned short* wz_bf   = (unsigned short*)alloc((size_t)V_DIM * HIDDEN * 2);
  unsigned short* wab_bf  = (unsigned short*)alloc((size_t)64 * HIDDEN * 2);
  unsigned short* wout_bf = (unsigned short*)alloc((size_t)HIDDEN * V_DIM * 2);
  float* qkv_raw  = (float*)alloc((size_t)NTOK * CONV_INNER * 4);
  float* qkv_conv = (float*)alloc((size_t)NTOK * CONV_INNER * 4);
  float* zbuf     = (float*)alloc((size_t)NTOK * V_DIM * 4);
  float* abbuf    = (float*)alloc((size_t)NTOK * 64 * 4);
  float* qn       = (float*)alloc((size_t)NTOK * QK_DIM * 4);
  float* kn       = (float*)alloc((size_t)NTOK * QK_DIM * 4);
  float* gbuf     = (float*)alloc((size_t)NTOK * NUM_V_HEADS * 4);
  float* betabuf  = (float*)alloc((size_t)NTOK * NUM_V_HEADS * 4);
  float* ybuf     = (float*)alloc((size_t)NTOK * V_DIM * 4);
  unsigned short* yout_bf = (unsigned short*)alloc((size_t)NTOK * V_DIM * 2);
  (void)ws_size;

  auto cast = [&](const float* src, unsigned short* dst, int n) {
    cast_bf16_kernel<<<(n + 255) / 256, 256, 0, stream>>>(src, dst, n);
  };
  cast(hidden, h_bf, NTOK * HIDDEN);
  cast(W_qkv, wqkv_bf, CONV_INNER * HIDDEN);
  cast(W_z, wz_bf, V_DIM * HIDDEN);
  cast(W_a, wab_bf, 32 * HIDDEN);
  cast(W_b, wab_bf + 32 * HIDDEN, 32 * HIDDEN);
  cast(W_out, wout_bf, HIDDEN * V_DIM);

  gemm_bf16_kernel<<<dim3(CONV_INNER / BN, NTOK / BM), 256, 0, stream>>>(
      h_bf, wqkv_bf, qkv_raw, CONV_INNER, HIDDEN);
  gemm_bf16_kernel<<<dim3(V_DIM / BN, NTOK / BM), 256, 0, stream>>>(
      h_bf, wz_bf, zbuf, V_DIM, HIDDEN);
  gemm_bf16_kernel<<<dim3(1, NTOK / BM), 256, 0, stream>>>(
      h_bf, wab_bf, abbuf, 64, HIDDEN);

  conv_silu_kernel<<<dim3(CONV_INNER / 256, NTOK), 256, 0, stream>>>(
      qkv_raw, conv_w, qkv_conv, conv_state);

  l2norm_kernel<<<NTOK * NUM_K_HEADS, 128, 0, stream>>>(qkv_conv, qn, kn);

  gbeta_kernel<<<(NTOK * NUM_V_HEADS) / 256, 256, 0, stream>>>(
      abbuf, A_log, dt_bias, gbuf, betabuf);

  scan_kernel<<<BATCH * NUM_V_HEADS, 256, 0, stream>>>(
      qn, kn, qkv_conv, gbuf, betabuf, ybuf, state_out);

  gate_kernel<<<NTOK * NUM_V_HEADS, 128, 0, stream>>>(ybuf, zbuf, norm_w, yout_bf);

  gemm_bf16_kernel<<<dim3(HIDDEN / BN, NTOK / BM), 256, 0, stream>>>(
      yout_bf, wout_bf, out, HIDDEN, V_DIM);
}
